// MultiHeadAttention_58420145160770
// MI455X (gfx1250) — matrix operand order repaired and checked by reading, not compiled, not run on hardware
//
#include <hip/hip_runtime.h>
#include <hip/hip_bf16.h>

// ---------------------------------------------------------------------------
// Feature probes (false in host pass -> safe fallbacks everywhere)
// ---------------------------------------------------------------------------
#if defined(__AMDGCN__) && __has_builtin(__builtin_amdgcn_tensor_load_to_lds)
#define HAVE_TDM 1
#endif
#if defined(__AMDGCN__) && __has_builtin(__builtin_amdgcn_global_load_async_to_lds_b128)
#define HAVE_ASYNC 1
#endif

// ---------------------------------------------------------------------------
// Types
// ---------------------------------------------------------------------------
typedef __attribute__((ext_vector_type(16))) __bf16       v16bf;
typedef __attribute__((ext_vector_type(8)))  float        v8f;
typedef __attribute__((ext_vector_type(4)))  unsigned int u32x4;
typedef __attribute__((ext_vector_type(2)))  unsigned int u32x2;
typedef __attribute__((ext_vector_type(4)))  int          i32x4;
typedef __attribute__((ext_vector_type(8)))  int          tdm_i32x8;
typedef __attribute__((ext_vector_type(4)))  int          tdm_i32x4;
typedef __attribute__((ext_vector_type(4)))  unsigned int tdm_u32x4;

// AS-qualified pointer types for the async-to-LDS builtin
typedef __attribute__((address_space(1))) i32x4* gas_v4i;
typedef __attribute__((address_space(3))) i32x4* las_v4i;

union FragAB {                 // 16 bf16 = 8 VGPRs (A or B operand)
    v16bf          v;
    u32x4          q[2];
    unsigned short u[16];
};
union FragC {                  // 8 f32 accumulator
    v8f   v;
    float f[8];
};

__device__ __forceinline__ unsigned short f32_to_bf16(float f) {
    unsigned int u = __float_as_uint(f);
    u += 0x7FFFu + ((u >> 16) & 1u);   // round to nearest even
    return (unsigned short)(u >> 16);
}

__device__ __forceinline__ float half16_reduce_max(float v) {
    #pragma unroll
    for (int m = 1; m < 16; m <<= 1) v = fmaxf(v, __shfl_xor(v, m, 32));
    return v;
}
__device__ __forceinline__ float half16_reduce_sum(float v) {
    #pragma unroll
    for (int m = 1; m < 16; m <<= 1) v += __shfl_xor(v, m, 32);
    return v;
}

// ---------------------------------------------------------------------------
// Global -> LDS copy primitives
// ---------------------------------------------------------------------------
// Per-lane 16B async copy (ASYNCcnt) or sync fallback.
__device__ __forceinline__ void cp16(void* lds, const void* g) {
#if defined(HAVE_ASYNC)
    __builtin_amdgcn_global_load_async_to_lds_b128(
        (gas_v4i)(unsigned long long)(size_t)g,
        (las_v4i)(unsigned int)(size_t)lds,
        0, 0);
#else
    *(u32x4*)lds = *(const u32x4*)g;
#endif
}
__device__ __forceinline__ void cp_wait() {
#if defined(HAVE_ASYNC)
    asm volatile("s_wait_asynccnt 0x0" ::: "memory");
#endif
}

#if defined(HAVE_TDM)
// TDM 2D tile load: bf16 elements, tile (tile_w x tile_h), row stride in
// elements, LDS padding pad_amount DWORD-code after every pad_interval-code
// DWORDs (ISA 08 sec 8.4).  Issued once per wave; tracked by TENSORcnt.
__device__ __forceinline__ void tdm_load_tile_bf16(
    unsigned int lds_byte_addr, const void* gptr,
    unsigned int tile_w, unsigned int tile_h,
    unsigned int tensor_w, unsigned int tensor_h,
    unsigned long long row_stride_elems,
    unsigned int pad_interval_code, unsigned int pad_amount_code) {
    union { unsigned long long q[2]; tdm_u32x4 v; } g0;
    union { unsigned long long q[4]; tdm_i32x8 v; } g1;
    union { unsigned long long q[2]; tdm_i32x4 v; } gz;
    unsigned long long ga = (unsigned long long)(size_t)gptr;
    g0.q[0] = 1ull                                            // count = 1
            | ((unsigned long long)lds_byte_addr << 32);      // lds_addr
    g0.q[1] = (ga & 0x01FFFFFFFFFFFFFFull)                    // global_addr[56:0]
            | (2ull << 62);                                   // type = 2 (image)
    g1.q[0] = (1ull << 16)                                    // data_size = 2B
            | (1ull << 20)                                    // pad_enable
            | ((unsigned long long)(pad_interval_code & 7u) << 22)
            | ((unsigned long long)(pad_amount_code & 127u) << 25)
            | ((unsigned long long)(tensor_w & 0xFFFFu) << 48);   // dim0[15:0]
    g1.q[1] = ((unsigned long long)(tensor_w >> 16) & 0xFFFFull)  // dim0[31:16]
            | ((unsigned long long)tensor_h << 16)                // dim1
            | ((unsigned long long)(tile_w & 0xFFFFu) << 48);     // tile_dim0
    g1.q[2] = (unsigned long long)(tile_h & 0xFFFFu)              // tile_dim1
            | ((row_stride_elems & 0xFFFFFFFFull) << 32);         // stride[31:0]
    g1.q[3] = (row_stride_elems >> 32) & 0xFFFFull;               // stride[47:32]
    gz.q[0] = 0; gz.q[1] = 0;
#if __clang_major__ >= 23
    union { unsigned long long q[4]; tdm_i32x8 v; } gz8;
    gz8.q[0] = gz8.q[1] = gz8.q[2] = gz8.q[3] = 0;
    __builtin_amdgcn_tensor_load_to_lds(g0.v, g1.v, gz.v, gz.v, gz8.v, 0);
#else
    __builtin_amdgcn_tensor_load_to_lds(g0.v, g1.v, gz.v, gz.v, 0);
#endif
}
#endif

// ---------------------------------------------------------------------------
// Problem constants
// ---------------------------------------------------------------------------
#define BB 2
#define TT 2048
#define DD 1024
#define HH 16
#define HD 64

// ---------------------------------------------------------------------------
// fp32 -> bf16 conversion (4 floats / thread)
// ---------------------------------------------------------------------------
__global__ __launch_bounds__(256) void cvt_bf16(const float* __restrict__ s,
                                                unsigned short* __restrict__ d,
                                                int n4) {
    int i = blockIdx.x * 256 + threadIdx.x;
    if (i < n4) {
        float4 f = ((const float4*)s)[i];
        u32x2 o;
        o.x = (unsigned)f32_to_bf16(f.x) | ((unsigned)f32_to_bf16(f.y) << 16);
        o.y = (unsigned)f32_to_bf16(f.z) | ((unsigned)f32_to_bf16(f.w) << 16);
        ((u32x2*)d)[i] = o;
    }
}

// ---------------------------------------------------------------------------
// WMMA GEMM:  C[m,n] = sum_k A[m,k] * Bw[n,k]   (y = x @ W^T)
// Double-buffered LDS slabs; next slab staged by TDM (or async-to-LDS) while
// WMMAs consume the current one.
// MODE 0: fp32 C -> Cout[M,N];  MODE 1: fused RoPE + q/k/v split-store.
// ---------------------------------------------------------------------------
template <int MODE>
__global__ __launch_bounds__(256) void gemm_bf16(
    const unsigned short* __restrict__ A,    // [M,K] bf16
    const unsigned short* __restrict__ Bw,   // [N,K] bf16
    float* __restrict__ Cout,
    const float* __restrict__ cosT,
    const float* __restrict__ sinT,
    unsigned short* __restrict__ qb,
    unsigned short* __restrict__ kb,
    unsigned short* __restrict__ vb,
    int M, int N, int K) {

    const int tid  = threadIdx.x;
    const int lane = tid & 31;
    const int wave = tid >> 5;
    const int wm   = wave >> 1;          // 0..3
    const int wn   = wave & 1;           // 0..1
    const int bm   = blockIdx.y * 128;
    const int bn   = blockIdx.x * 128;
    const int l15  = lane & 15;

    __shared__ __align__(16) unsigned short As[2][128][40];   // 64B data + 16B pad
    __shared__ __align__(16) unsigned short Bs[2][128][40];

    FragC acc[2][4];
    #pragma unroll
    for (int mi = 0; mi < 2; ++mi)
        #pragma unroll
        for (int nf = 0; nf < 4; ++nf)
            #pragma unroll
            for (int j = 0; j < 8; ++j) acc[mi][nf].f[j] = 0.0f;

    const int lrow = tid >> 1;           // 0..127
    const int lcol = (tid & 1) * 16;     // 0 / 16
    const int hk   = (lane < 16) ? 0 : 8;
    const int kb0  = (lane < 16) ? 0 : 8;

    const auto issue_slab = [&](int k0, int buf) {
#if defined(HAVE_TDM)
        if (wave == 0) {
            // pad_interval 3 -> 16 DWORDs (64B row), pad_amount 3 -> 4 DWORDs
            tdm_load_tile_bf16((unsigned int)(size_t)&As[buf][0][0],
                               A + (size_t)bm * K + k0, 32, 128,
                               (unsigned)K, 128, (unsigned long long)K, 3, 3);
            tdm_load_tile_bf16((unsigned int)(size_t)&Bs[buf][0][0],
                               Bw + (size_t)bn * K + k0, 32, 128,
                               (unsigned)K, 128, (unsigned long long)K, 3, 3);
        }
#else
        const unsigned short* ga = A  + (size_t)(bm + lrow) * K + k0 + lcol;
        const unsigned short* gb = Bw + (size_t)(bn + lrow) * K + k0 + lcol;
        cp16(&As[buf][lrow][lcol],     ga);
        cp16(&As[buf][lrow][lcol + 8], ga + 8);
        cp16(&Bs[buf][lrow][lcol],     gb);
        cp16(&Bs[buf][lrow][lcol + 8], gb + 8);
#endif
    };
    const auto wait_slab = [&] {
#if defined(HAVE_TDM)
        if (wave == 0) __builtin_amdgcn_s_wait_tensorcnt(0);
#else
        cp_wait();
#endif
    };

    issue_slab(0, 0);
    wait_slab();
    __syncthreads();

    int buf = 0;
    for (int k0 = 0; k0 < K; k0 += 32) {
        if (k0 + 32 < K) issue_slab(k0 + 32, buf ^ 1);   // overlap with WMMAs

        FragAB bfr[4];
        #pragma unroll
        for (int nf = 0; nf < 4; ++nf) {
            const int n0 = wn * 64 + nf * 16 + l15;
            bfr[nf].q[0] = *(const u32x4*)&Bs[buf][n0][kb0];
            bfr[nf].q[1] = *(const u32x4*)&Bs[buf][n0][kb0 + 16];
        }
        #pragma unroll
        for (int mi = 0; mi < 2; ++mi) {
            const int m0 = wm * 32 + mi * 16 + l15;
            FragAB af;
            af.q[0] = *(const u32x4*)&As[buf][m0][hk];
            af.q[1] = *(const u32x4*)&As[buf][m0][hk + 16];
            #pragma unroll
            for (int nf = 0; nf < 4; ++nf)
                acc[mi][nf].v = __builtin_amdgcn_wmma_f32_16x16x32_bf16(
                    false, af.v, false, bfr[nf].v, (short)0, acc[mi][nf].v,
                    false, false);
        }

        wait_slab();
        __syncthreads();
        buf ^= 1;
    }

    if (MODE == 0) {
        #pragma unroll
        for (int mi = 0; mi < 2; ++mi)
            #pragma unroll
            for (int nf = 0; nf < 4; ++nf)
                #pragma unroll
                for (int r = 0; r < 8; ++r) {
                    int row = bm + wm * 32 + mi * 16 + ((lane < 16) ? r : r + 8);
                    int col = bn + wn * 64 + nf * 16 + l15;
                    Cout[(size_t)row * N + col] = acc[mi][nf].f[r];
                }
    } else {
        const int colbase = bn + wn * 64;          // multiple of 64
        const int sel     = colbase >> 10;         // 0=q,1=k,2=v
        const int h       = (colbase >> 6) & (HH - 1);
        unsigned short* dst = (sel == 0) ? qb : ((sel == 1) ? kb : vb);
        #pragma unroll
        for (int mi = 0; mi < 2; ++mi) {
            #pragma unroll
            for (int r = 0; r < 8; ++r) {
                const int row = bm + wm * 32 + mi * 16 + ((lane < 16) ? r : r + 8);
                const int t   = row & (TT - 1);
                const int b   = row >> 11;
                const size_t base = ((size_t)(b * HH + h) * TT + t) * HD;
                if (sel < 2) {   // q,k : RoPE
                    #pragma unroll
                    for (int p = 0; p < 2; ++p) {
                        const int hd_lo = p * 16 + l15;      // < 32
                        const int hd_hi = hd_lo + 32;
                        float c_lo = cosT[t * HD + hd_lo];
                        float s_lo = sinT[t * HD + hd_lo];
                        float c_hi = cosT[t * HD + hd_hi];
                        float s_hi = sinT[t * HD + hd_hi];
                        float xlo = acc[mi][p].f[r];
                        float xhi = acc[mi][p + 2].f[r];
                        float olo = xlo * c_lo - xhi * s_lo;
                        float ohi = xhi * c_hi + xlo * s_hi;
                        dst[base + hd_lo] = f32_to_bf16(olo);
                        dst[base + hd_hi] = f32_to_bf16(ohi);
                    }
                } else {         // v : passthrough
                    #pragma unroll
                    for (int nf = 0; nf < 4; ++nf)
                        dst[base + nf * 16 + l15] = f32_to_bf16(acc[mi][nf].f[r]);
                }
            }
        }
    }
}

// ---------------------------------------------------------------------------
// Flash attention, causal. One block per (b, h, 128-row query tile).
// 8 waves x 16 query rows; K tile staged async-to-LDS, V transposed in-flight.
// ---------------------------------------------------------------------------
__global__ __launch_bounds__(256) void attn_kernel(
    const unsigned short* __restrict__ qb,
    const unsigned short* __restrict__ kb,
    const unsigned short* __restrict__ vb,
    unsigned short* __restrict__ ob) {

    const int lane = threadIdx.x & 31;
    const int w    = threadIdx.x >> 5;       // wave 0..7
    const int qt   = blockIdx.x;             // 128-row query tile
    const int h    = blockIdx.y;
    const int b    = blockIdx.z;
    const int l15  = lane & 15;
    const size_t bh = (size_t)(b * HH + h) * TT;

    __shared__ __align__(16) unsigned short Kt [64][72];   // [key][d]
    __shared__ __align__(16) unsigned short VtT[64][72];   // [d][key]
    __shared__ __align__(16) unsigned short Ps [8][16][72];

    const int qrow0 = qt * 128 + w * 16;
    const int hk    = (lane < 16) ? 0 : 8;    // A-frag K base
    const int kbs   = (lane < 16) ? 0 : 8;    // B-frag K base

    FragAB aq[2];
    {
        const unsigned short* qp = qb + (bh + qrow0 + l15) * HD;
        aq[0].q[0] = *(const u32x4*)(qp + hk);
        aq[0].q[1] = *(const u32x4*)(qp + hk + 16);
        aq[1].q[0] = *(const u32x4*)(qp + 32 + hk);
        aq[1].q[1] = *(const u32x4*)(qp + 32 + hk + 16);
    }

    float rowmax[8], rowsum[8];
    FragC accO[4];
    #pragma unroll
    for (int r = 0; r < 8; ++r) { rowmax[r] = -1e30f; rowsum[r] = 0.0f; }
    #pragma unroll
    for (int nf = 0; nf < 4; ++nf)
        #pragma unroll
        for (int j = 0; j < 8; ++j) accO[nf].f[j] = 0.0f;

    const int lrow = threadIdx.x >> 2;        // 0..63
    const int lcol = (threadIdx.x & 3) * 16;  // 0/16/32/48

    const int ktmax = 2 * qt + 1;             // causal: keys <= q max
    for (int kt = 0; kt <= ktmax; ++kt) {
        __syncthreads();
        {
            const size_t krow = bh + kt * 64 + lrow;
            cp16(&Kt[lrow][lcol],     kb + krow * HD + lcol);
            cp16(&Kt[lrow][lcol + 8], kb + krow * HD + lcol + 8);
            union { u32x4 q[2]; unsigned short u[16]; } vt;
            const u32x4* vp = (const u32x4*)(vb + krow * HD + lcol);
            vt.q[0] = vp[0]; vt.q[1] = vp[1];
            #pragma unroll
            for (int j = 0; j < 16; ++j) VtT[lcol + j][lrow] = vt.u[j];
            if (kt < ktmax) {     // pull next tile toward GL2
                __builtin_prefetch(kb + (krow + 64) * HD + lcol, 0, 0);
                __builtin_prefetch(vb + (krow + 64) * HD + lcol, 0, 0);
            }
        }
        cp_wait();
        __syncthreads();

        // S = Q K^T   (16 q-rows x 64 keys per wave)
        FragC s[4];
        #pragma unroll
        for (int nf = 0; nf < 4; ++nf) {
            #pragma unroll
            for (int j = 0; j < 8; ++j) s[nf].f[j] = 0.0f;
            const int n0 = nf * 16 + l15;
            #pragma unroll
            for (int st = 0; st < 2; ++st) {
                FragAB bfr;
                bfr.q[0] = *(const u32x4*)&Kt[n0][st * 32 + kbs];
                bfr.q[1] = *(const u32x4*)&Kt[n0][st * 32 + kbs + 16];
                s[nf].v = __builtin_amdgcn_wmma_f32_16x16x32_bf16(
                    false, aq[st].v, false, bfr.v, (short)0, s[nf].v,
                    false, false);
            }
        }

        // scale + causal mask (only tiles that touch the diagonal)
        const bool needmask = (kt * 64 + 63 > qrow0);
        #pragma unroll
        for (int nf = 0; nf < 4; ++nf)
            #pragma unroll
            for (int r = 0; r < 8; ++r) {
                float sv = s[nf].f[r] * 0.125f;   // 1/sqrt(64)
                if (needmask) {
                    int kg = kt * 64 + nf * 16 + l15;
                    int qg = qrow0 + ((lane < 16) ? r : r + 8);
                    if (kg > qg) sv = -1e30f;
                }
                s[nf].f[r] = sv;
            }

        // online softmax
        #pragma unroll
        for (int r = 0; r < 8; ++r) {
            float tm = fmaxf(fmaxf(s[0].f[r], s[1].f[r]),
                             fmaxf(s[2].f[r], s[3].f[r]));
            tm = half16_reduce_max(tm);
            float mnew  = fmaxf(rowmax[r], tm);
            float alpha = __expf(rowmax[r] - mnew);
            rowmax[r] = mnew;
            float ts = 0.0f;
            #pragma unroll
            for (int nf = 0; nf < 4; ++nf) {
                float p = __expf(s[nf].f[r] - mnew);
                s[nf].f[r] = p;
                ts += p;
            }
            ts = half16_reduce_sum(ts);
            rowsum[r] = rowsum[r] * alpha + ts;
            #pragma unroll
            for (int nf = 0; nf < 4; ++nf) accO[nf].f[r] *= alpha;
        }

        // P: C-layout -> A-layout via per-wave LDS scratch
        #pragma unroll
        for (int nf = 0; nf < 4; ++nf)
            #pragma unroll
            for (int r = 0; r < 8; ++r) {
                int rr = (lane < 16) ? r : r + 8;
                Ps[w][rr][nf * 16 + l15] = f32_to_bf16(s[nf].f[r]);
            }
        asm volatile("s_wait_dscnt 0x0" ::: "memory");
        FragAB ap[2];
        ap[0].q[0] = *(const u32x4*)&Ps[w][l15][hk];
        ap[0].q[1] = *(const u32x4*)&Ps[w][l15][hk + 16];
        ap[1].q[0] = *(const u32x4*)&Ps[w][l15][32 + hk];
        ap[1].q[1] = *(const u32x4*)&Ps[w][l15][32 + hk + 16];

        // O += P V
        #pragma unroll
        for (int nf = 0; nf < 4; ++nf) {
            const int n0 = nf * 16 + l15;
            #pragma unroll
            for (int st = 0; st < 2; ++st) {
                FragAB bfr;
                bfr.q[0] = *(const u32x4*)&VtT[n0][st * 32 + kbs];
                bfr.q[1] = *(const u32x4*)&VtT[n0][st * 32 + kbs + 16];
                accO[nf].v = __builtin_amdgcn_wmma_f32_16x16x32_bf16(
                    false, ap[st].v, false, bfr.v, (short)0, accO[nf].v,
                    false, false);
            }
        }
    }

    // normalize + store as [B, T, H, HD] (== [B*T, D] row-major)
    float inv[8];
    #pragma unroll
    for (int r = 0; r < 8; ++r) inv[r] = 1.0f / rowsum[r];
    #pragma unroll
    for (int nf = 0; nf < 4; ++nf)
        #pragma unroll
        for (int r = 0; r < 8; ++r) {
            int row = qrow0 + ((lane < 16) ? r : r + 8);
            int hd  = nf * 16 + l15;
            ob[((size_t)(b * TT + row) * HH + h) * HD + hd] =
                f32_to_bf16(accO[nf].f[r] * inv[r]);
        }
}

// ---------------------------------------------------------------------------
// Launch
// ---------------------------------------------------------------------------
extern "C" void kernel_launch(void* const* d_in, const int* in_sizes, int n_in,
                              void* d_out, int out_size, void* d_ws, size_t ws_size,
                              hipStream_t stream) {
    const float* x    = (const float*)d_in[0];
    const float* cosT = (const float*)d_in[1];
    const float* sinT = (const float*)d_in[2];
    // d_in[3] = additive causal mask: applied analytically (identical result)
    const float* Wq = (const float*)d_in[4];
    const float* Wk = (const float*)d_in[5];
    const float* Wv = (const float*)d_in[6];
    const float* Wo = (const float*)d_in[7];
    float* out = (float*)d_out;

    const size_t NX = (size_t)BB * TT * DD;        // 4,194,304
    const size_t NW = (size_t)DD * DD;             // 1,048,576

    unsigned short* xb   = (unsigned short*)d_ws;
    unsigned short* wcat = xb + NX;                // [3072,1024]
    unsigned short* wo   = wcat + 3 * NW;
    unsigned short* qbuf = wo + NW;                // [B,H,T,HD]
    unsigned short* kbuf = qbuf + NX;
    unsigned short* vbuf = kbuf + NX;
    unsigned short* obuf = vbuf + NX;              // [B,T,H,HD]

    cvt_bf16<<<(int)(NX / 4 + 255) / 256, 256, 0, stream>>>(x, xb, (int)(NX / 4));
    cvt_bf16<<<(int)(NW / 4 + 255) / 256, 256, 0, stream>>>(Wq, wcat,          (int)(NW / 4));
    cvt_bf16<<<(int)(NW / 4 + 255) / 256, 256, 0, stream>>>(Wk, wcat + NW,     (int)(NW / 4));
    cvt_bf16<<<(int)(NW / 4 + 255) / 256, 256, 0, stream>>>(Wv, wcat + 2 * NW, (int)(NW / 4));
    cvt_bf16<<<(int)(NW / 4 + 255) / 256, 256, 0, stream>>>(Wo, wo,            (int)(NW / 4));

    // fused QKV GEMM + RoPE:  [4096,1024] x [3072,1024]^T
    gemm_bf16<1><<<dim3(3072 / 128, 4096 / 128), 256, 0, stream>>>(
        xb, wcat, nullptr, cosT, sinT, qbuf, kbuf, vbuf, 4096, 3072, 1024);

    // causal flash attention
    attn_kernel<<<dim3(TT / 128, HH, BB), 256, 0, stream>>>(qbuf, kbuf, vbuf, obuf);

    // output projection: [4096,1024] x [1024,1024]^T -> fp32 d_out
    gemm_bf16<0><<<dim3(1024 / 128, 4096 / 128), 256, 0, stream>>>(
        obuf, wo, out, nullptr, nullptr, nullptr, nullptr, nullptr,
        4096, 1024, 1024);
}
